// BinSAGE_45921790329541
// MI455X (gfx1250) — compile-verified, matrix operand order checked by reading, not run on hardware
//
#include <hip/hip_runtime.h>
#include <stdint.h>
#include <stddef.h>

// BinSAGE on MI455X (gfx1250): sparse mean-aggregation via f32 global atomics
// (bandwidth-bound, entire working set ~61MB fits in the 192MB L2), dense
// projections via v_wmma_f32_16x16x32_bf16 with fused bias/ReLU epilogues.
// K is now a compile-time template parameter -> straight-line WMMA pipeline
// (no per-step s_cmp/s_cbranch), constant address offsets folded into the
// global_load_b128 clauses.

typedef __bf16 bf16;
typedef __attribute__((ext_vector_type(16))) bf16  v16bf;
typedef __attribute__((ext_vector_type(8)))  bf16  v8bf;
typedef __attribute__((ext_vector_type(8)))  float v8f;

#define N_NODES 50000
#define IN_DIM  96
#define HID     128
#define OUT_DIM 64

// ---------------- elementwise / prep kernels ----------------

__global__ void k_zero_f32(float* __restrict__ p, int n) {
  int i = blockIdx.x * blockDim.x + threadIdx.x;
  if (i < n) p[i] = 0.0f;
}

__global__ void k_cvt_bf16(const float* __restrict__ s, bf16* __restrict__ d, int n) {
  int i = blockIdx.x * blockDim.x + threadIdx.x;
  if (i < n) d[i] = (bf16)s[i];
}

// straight-through sign binarization (forward path only): sign(0) == 0
__global__ void k_sign_bf16(const float* __restrict__ s, bf16* __restrict__ d, int n) {
  int i = blockIdx.x * blockDim.x + threadIdx.x;
  if (i < n) {
    float w = s[i];
    float r = (w > 0.0f) ? 1.0f : ((w < 0.0f) ? -1.0f : 0.0f);
    d[i] = (bf16)r;
  }
}

// ---------------- graph kernels ----------------

__global__ void k_degree(const int* __restrict__ didx, float* __restrict__ deg, int E) {
  int e = blockIdx.x * blockDim.x + threadIdx.x;
  if (e < E) atomicAdd(deg + didx[e], 1.0f);
}

// one thread = (edge, 8-feature chunk): 16B bf16 gather + 8 f32 atomic adds
__global__ void k_scatter8(const bf16* __restrict__ feat, const int* __restrict__ sidx,
                           const int* __restrict__ didx, float* __restrict__ agg,
                           int E, int F) {
  int e = blockIdx.x * blockDim.x + threadIdx.x;
  if (e >= E) return;
  int ch = blockIdx.y << 3;
  int s = sidx[e];
  int d = didx[e];
  v8bf v = *(const v8bf*)(feat + (size_t)s * F + ch);
  float* ap = agg + (size_t)d * F + ch;
#pragma unroll
  for (int i = 0; i < 8; ++i) atomicAdd(ap + i, (float)v[i]);
}

// mean = sum / max(deg,1); emit bf16 operand buffer for WMMA
__global__ void k_mean_bf16(const float* __restrict__ agg, const float* __restrict__ deg,
                            bf16* __restrict__ out, int F) {
  int row = blockIdx.x;
  int f   = threadIdx.x;
  float dv = fmaxf(deg[row], 1.0f);
  size_t idx = (size_t)row * F + f;
  out[idx] = (bf16)(agg[idx] / dv);
}

// ---------------- WMMA GEMM ----------------
// A fragment (16x32 bf16, ISA 7.12.2): lane<16 -> row=lane, K {0..7}u{16..23};
// lane>=16 -> row=lane-16, K {8..15}u{24..31}.  Two contiguous 16B chunks/lane.
// B (32x16) mirrors this with col = lane%16 when weights are stored [n][k].
__device__ __forceinline__ v16bf load_frag(const bf16* __restrict__ p) {
  v8bf lo = *(const v8bf*)(p);
  v8bf hi = *(const v8bf*)(p + 16);
  return __builtin_shufflevector(lo, hi, 0, 1, 2, 3, 4, 5, 6, 7,
                                 8, 9, 10, 11, 12, 13, 14, 15);
}

// out[m][n] = act( bias[n] + A1[m,:]*W1[n,:] + A2[m,:]*W2[n,:] )
// grid.x = M/16 tiles; blockDim.x = (Nout/16)*32, one wave per 16x16 N-tile.
// K, Nout compile-time -> fully unrolled straight-line WMMA pipeline.
template <int K, int Nout, bool RELU, bool STORE_BF16>
__global__ void __launch_bounds__(256)
k_sage_gemm(const bf16* __restrict__ A1, const bf16* __restrict__ A2,
            const bf16* __restrict__ W1, const bf16* __restrict__ W2,
            const float* __restrict__ bias,
            bf16* __restrict__ outb, float* __restrict__ outf) {
  const int lane    = threadIdx.x & 31;
  const int ntile   = threadIdx.x >> 5;
  const int row0    = blockIdx.x << 4;
  const int m_local = lane & 15;
  const int k_half  = (lane >> 4) << 3;        // 0 or 8

  const bf16* a1p = A1 + (size_t)(row0 + m_local) * K + k_half;
  const bf16* a2p = A2 + (size_t)(row0 + m_local) * K + k_half;
  const bf16* w1p = W1 + (size_t)((ntile << 4) + m_local) * K + k_half;
  const bf16* w2p = W2 + (size_t)((ntile << 4) + m_local) * K + k_half;

  // prefetch next M-tile's A rows (gfx1250 global_prefetch_b8)
  __builtin_prefetch(a1p + (size_t)16 * K, 0, 1);
  __builtin_prefetch(a2p + (size_t)16 * K, 0, 1);

  v8f c = {};
#pragma unroll
  for (int kb = 0; kb < K; kb += 32) {
    v16bf a = load_frag(a1p + kb);
    v16bf b = load_frag(w1p + kb);
    c = __builtin_amdgcn_wmma_f32_16x16x32_bf16(false, a, false, b,
                                                (short)0, c, false, false);
  }
#pragma unroll
  for (int kb = 0; kb < K; kb += 32) {
    v16bf a = load_frag(a2p + kb);
    v16bf b = load_frag(w2p + kb);
    c = __builtin_amdgcn_wmma_f32_16x16x32_bf16(false, a, false, b,
                                                (short)0, c, false, false);
  }

  // C/D layout: VGPR r -> row = row0 + r + 8*(lane>>4), col = ntile*16 + lane%16
  const int col   = (ntile << 4) + m_local;
  const int rbase = row0 + ((lane >> 4) << 3);
  const float bv  = bias[col];
#pragma unroll
  for (int r = 0; r < 8; ++r) {
    float v = c[r] + bv;
    if (RELU) v = fmaxf(v, 0.0f);
    size_t o = (size_t)(rbase + r) * Nout + col;
    if (STORE_BF16) outb[o] = (bf16)v;
    else            outf[o] = v;
  }
}

// ---------------- launcher ----------------

extern "C" void kernel_launch(void* const* d_in, const int* in_sizes, int n_in,
                              void* d_out, int out_size, void* d_ws, size_t ws_size,
                              hipStream_t stream) {
  (void)n_in; (void)out_size; (void)ws_size;
  const float* x    = (const float*)d_in[0];   // [50000 x 96]
  const int*   eidx = (const int*)d_in[1];     // [2 x E]
  const float* w1l  = (const float*)d_in[2];   // [128 x 96]
  const float* b1   = (const float*)d_in[3];   // [128]
  const float* w1r  = (const float*)d_in[4];   // [128 x 96]
  const float* w2l  = (const float*)d_in[5];   // [64 x 128]
  const float* b2   = (const float*)d_in[6];   // [64]
  const float* w2r  = (const float*)d_in[7];   // [64 x 128]

  const int E = in_sizes[1] / 2;
  const int* sidx = eidx;
  const int* didx = eidx + E;

  // workspace layout (256B-aligned offsets), ~61 MB total
  char* ws = (char*)d_ws;
  bf16*  xb   = (bf16*)(ws + 0);          //  9.6 MB  x in bf16
  bf16*  hb   = (bf16*)(ws + 9600000);    // 12.8 MB  layer-1 activations (bf16)
  float* aggF = (float*)(ws + 22400000);  // 25.6 MB  f32 scatter accumulator
  bf16*  aggB = (bf16*)(ws + 48000000);   // 12.8 MB  mean-agg in bf16
  float* deg  = (float*)(ws + 60800000);  //  0.2 MB
  bf16*  w1lb = (bf16*)(ws + 61000192);
  bf16*  w1rb = w1lb + HID * IN_DIM;
  bf16*  w2lb = w1rb + HID * IN_DIM;
  bf16*  w2rb = w2lb + OUT_DIM * HID;

  auto cdiv = [](int a, int b) { return (a + b - 1) / b; };

  // prep: x -> bf16, binarized weights -> bf16
  k_cvt_bf16<<<cdiv(N_NODES * IN_DIM, 256), 256, 0, stream>>>(x, xb, N_NODES * IN_DIM);
  k_sign_bf16<<<cdiv(HID * IN_DIM, 256), 256, 0, stream>>>(w1l, w1lb, HID * IN_DIM);
  k_sign_bf16<<<cdiv(HID * IN_DIM, 256), 256, 0, stream>>>(w1r, w1rb, HID * IN_DIM);
  k_sign_bf16<<<cdiv(OUT_DIM * HID, 256), 256, 0, stream>>>(w2l, w2lb, OUT_DIM * HID);
  k_sign_bf16<<<cdiv(OUT_DIM * HID, 256), 256, 0, stream>>>(w2r, w2rb, OUT_DIM * HID);

  // in-degrees (shared by both layers)
  k_zero_f32<<<cdiv(N_NODES, 256), 256, 0, stream>>>(deg, N_NODES);
  k_degree<<<cdiv(E, 256), 256, 0, stream>>>(didx, deg, E);

  // ---- layer 1 ----
  k_zero_f32<<<cdiv(N_NODES * IN_DIM, 256), 256, 0, stream>>>(aggF, N_NODES * IN_DIM);
  {
    dim3 g(cdiv(E, 256), IN_DIM / 8);
    k_scatter8<<<g, 256, 0, stream>>>(xb, sidx, didx, aggF, E, IN_DIM);
  }
  k_mean_bf16<<<N_NODES, IN_DIM, 0, stream>>>(aggF, deg, aggB, IN_DIM);
  // h = relu(b1 + agg@sign(w1_l)^T + x@sign(w1_r)^T), bf16 out
  k_sage_gemm<IN_DIM, HID, true, true><<<N_NODES / 16, (HID / 16) * 32, 0, stream>>>(
      aggB, xb, w1lb, w1rb, b1, hb, nullptr);

  // ---- layer 2 ----
  k_zero_f32<<<cdiv(N_NODES * HID, 256), 256, 0, stream>>>(aggF, N_NODES * HID);
  {
    dim3 g(cdiv(E, 256), HID / 8);
    k_scatter8<<<g, 256, 0, stream>>>(hb, sidx, didx, aggF, E, HID);
  }
  k_mean_bf16<<<N_NODES, HID, 0, stream>>>(aggF, deg, aggB, HID);
  // out = b2 + agg@sign(w2_l)^T + h@sign(w2_r)^T, f32 out -> d_out
  k_sage_gemm<HID, OUT_DIM, false, false><<<N_NODES / 16, (OUT_DIM / 16) * 32, 0, stream>>>(
      aggB, hb, w2lb, w2rb, b2, nullptr, (float*)d_out);
}
